// CrossAttention_72593537237711
// MI455X (gfx1250) — compile-verified, matrix operand order checked by reading, not run on hardware
//
#include <hip/hip_runtime.h>

// ---------------------------------------------------------------------------
// CrossAttention block for MI455X (gfx1250), wave32 + WMMA f32_16x16x32_f16.
// Double-buffered LDS pipeline; GLOBAL_LOAD_ASYNC_TO_LDS_B128 for f16 tiles.
// ---------------------------------------------------------------------------

#define B_   8
#define TQ_  1024
#define TK_  1024
#define D_   1024
#define H_   16
#define HD_  64
#define LDH  40    // LDS row stride in halves (80B, 16B-aligned, bank-spread)
#define ASZ  (128 * LDH)
#define WSZ  (64 * LDH)

typedef __attribute__((ext_vector_type(16))) _Float16 v16h;
typedef __attribute__((ext_vector_type(8)))  _Float16 h8;
typedef __attribute__((ext_vector_type(4)))  _Float16 h4;
typedef __attribute__((ext_vector_type(8)))  float    v8f;

// ---- CDNA5 async copy: global -> LDS, 16B per lane, ASYNCcnt-tracked ------
__device__ inline void async_copy_b128(const void* gsrc, void* lds) {
  const unsigned loff = (unsigned)(size_t)lds;   // generic LDS ptr: offset in [31:0]
  asm volatile("global_load_async_to_lds_b128 %0, %1, off"
               :: "v"(loff), "v"(gsrc) : "memory");
}
__device__ inline void async_wait0() {
  asm volatile("s_wait_asynccnt 0" ::: "memory");
}

// ---- fragment load: CDNA5 16-bit A/B operand layout -----------------------
// lane<16 : halves 0-7 -> K 0-7,  halves 8-15 -> K 16-23
// lane>=16: halves 0-7 -> K 8-15, halves 8-15 -> K 24-31
__device__ inline v16h ld_frag16(const _Float16* __restrict__ rowp, int lane) {
  const int kb = (lane >> 4) << 3;
  const h8 lo = *(const h8*)(rowp + kb);
  const h8 hi = *(const h8*)(rowp + kb + 16);
  return __builtin_shufflevector(lo, hi, 0, 1, 2, 3, 4, 5, 6, 7,
                                 8, 9, 10, 11, 12, 13, 14, 15);
}

// ---- 128x64 CTA tile GEMM core (double-buffered) --------------------------
// acc += A(128xK) @ W(64xK)^T.  A: f32 (convert pipeline) or f16 (async copy).
// W: always f16, staged with async-to-LDS. 256 threads = 8 waves; wave w
// computes rows 16w..16w+15 x 64 cols (4 WMMAs / 32-wide k-step).
template <typename TA>
__device__ inline void mm_tile_128x64(const TA* __restrict__ A, int lda,
                                      const _Float16* __restrict__ W, int ldw,
                                      int K, _Float16* Ash, _Float16* Wsh,
                                      v8f* acc) {
  const int tid  = threadIdx.x;
  const int lane = tid & 31;
  const int wave = tid >> 5;
  const int wr = tid >> 2, wc = (tid & 3) << 3;          // W-copy coords

  // ---- prologue: stage k0 = 0 into buffer 0 ----
  if constexpr (sizeof(TA) == 4) {
#pragma unroll
    for (int i = 0; i < 4; ++i) {
      const int idx = tid + (i << 8);
      const int r = idx >> 3, c4 = (idx & 7) << 2;
      const float4 v = *(const float4*)((const float*)A + (size_t)r * lda + c4);
      h4 o = {(_Float16)v.x, (_Float16)v.y, (_Float16)v.z, (_Float16)v.w};
      *(h4*)(Ash + r * LDH + c4) = o;
    }
  } else {
#pragma unroll
    for (int i = 0; i < 2; ++i) {
      const int idx = tid + (i << 8);
      const int r = idx >> 2, c8 = (idx & 3) << 3;
      async_copy_b128((const _Float16*)A + (size_t)r * lda + c8,
                      Ash + r * LDH + c8);
    }
  }
  async_copy_b128(W + (size_t)wr * ldw + wc, Wsh + wr * LDH + wc);

  int cur = 0;
#pragma unroll 1
  for (int k0 = 0; k0 < K; k0 += 32) {
    _Float16* Ac = Ash + cur * ASZ;
    _Float16* Wc = Wsh + cur * WSZ;
    _Float16* An = Ash + (cur ^ 1) * ASZ;
    _Float16* Wn = Wsh + (cur ^ 1) * WSZ;
    const bool more = (k0 + 32) < K;

    async_wait0();       // our async copies into buf[cur] are done
    __syncthreads();     // everyone's stores/copies visible

    // kick off next tile (other buffer) under this tile's compute
    float4 areg[4];
    if (more) {
      async_copy_b128(W + (size_t)wr * ldw + k0 + 32 + wc, Wn + wr * LDH + wc);
      if constexpr (sizeof(TA) == 4) {
#pragma unroll
        for (int i = 0; i < 4; ++i) {
          const int idx = tid + (i << 8);
          const int r = idx >> 3, c4 = (idx & 7) << 2;
          areg[i] = *(const float4*)((const float*)A + (size_t)r * lda + k0 + 32 + c4);
        }
      } else {
#pragma unroll
        for (int i = 0; i < 2; ++i) {
          const int idx = tid + (i << 8);
          const int r = idx >> 2, c8 = (idx & 3) << 3;
          async_copy_b128((const _Float16*)A + (size_t)r * lda + k0 + 32 + c8,
                          An + r * LDH + c8);
        }
      }
    }

    // compute on buf[cur]: preload all B frags, then 4 back-to-back WMMAs
    const v16h a = ld_frag16(Ac + (wave * 16 + (lane & 15)) * LDH, lane);
    v16h bf[4];
#pragma unroll
    for (int j = 0; j < 4; ++j)
      bf[j] = ld_frag16(Wc + (j * 16 + (lane & 15)) * LDH, lane);
#pragma unroll
    for (int j = 0; j < 4; ++j)
      acc[j] = __builtin_amdgcn_wmma_f32_16x16x32_f16(false, a, false, bf[j],
                                                      (short)0, acc[j], false, false);

    if (more) {
      if constexpr (sizeof(TA) == 4) {
#pragma unroll
        for (int i = 0; i < 4; ++i) {
          const int idx = tid + (i << 8);
          const int r = idx >> 3, c4 = (idx & 7) << 2;
          h4 o = {(_Float16)areg[i].x, (_Float16)areg[i].y,
                  (_Float16)areg[i].z, (_Float16)areg[i].w};
          *(h4*)(An + r * LDH + c4) = o;
        }
      }
    }
    cur ^= 1;
  }
}

// ---- generic GEMM: C = epi(A @ W^T + bias) --------------------------------
// outmode: 0 = f32 row-major, 1 = f16 row-major, 2 = f16 V^T per-head
__global__ __launch_bounds__(256, 2)
void gemm_epi_kernel(const float* __restrict__ A, int lda,
                     const _Float16* __restrict__ W, int ldw,
                     const float* __restrict__ bias,
                     void* __restrict__ Cv, int ldc, int K, int epi, int outmode) {
  __shared__ alignas(16) _Float16 Ash[2 * ASZ];
  __shared__ alignas(16) _Float16 Wsh[2 * WSZ];
  const int m0 = blockIdx.y * 128;
  const int n0 = blockIdx.x * 64;
  v8f acc[4];
#pragma unroll
  for (int j = 0; j < 4; ++j)
#pragma unroll
    for (int i = 0; i < 8; ++i) acc[j][i] = 0.0f;

  mm_tile_128x64<float>(A + (size_t)m0 * lda, lda, W + (size_t)n0 * ldw, ldw,
                        K, Ash, Wsh, acc);

  const int lane = threadIdx.x & 31;
  const int wave = threadIdx.x >> 5;
  const int mrow = m0 + wave * 16 + ((lane >> 4) << 3);
#pragma unroll
  for (int j = 0; j < 4; ++j) {
    const int n = n0 + j * 16 + (lane & 15);
    const float bv = bias ? bias[n] : 0.0f;
    if (outmode == 2) {
      h8 pk;
#pragma unroll
      for (int i = 0; i < 8; ++i) pk[i] = (_Float16)(acc[j][i] + bv);
      const size_t bh = (size_t)(mrow >> 10) * H_ + (n >> 6);
      _Float16* dst = (_Float16*)Cv + (bh * HD_ + (n & 63)) * TK_ + (mrow & 1023);
      *(h8*)dst = pk;
    } else {
#pragma unroll
      for (int i = 0; i < 8; ++i) {
        float v = acc[j][i] + bv;
        if (epi == 1) v = v * (1.0f / (1.0f + __expf(-v)));  // SiLU
        if (outmode == 1)
          ((_Float16*)Cv)[(size_t)(mrow + i) * ldc + n] = (_Float16)v;
        else
          ((float*)Cv)[(size_t)(mrow + i) * ldc + n] = v;
      }
    }
  }
}

// ---- attention scores: S[b,h,q,k] = (Q . K) / 8, Q/K f16 (all-async) ------
__global__ __launch_bounds__(256, 2)
void scores_kernel(const _Float16* __restrict__ Q, const _Float16* __restrict__ Kf,
                   float* __restrict__ S) {
  __shared__ alignas(16) _Float16 Ash[2 * ASZ];
  __shared__ alignas(16) _Float16 Wsh[2 * WSZ];
  const int z = blockIdx.z, b = z >> 4, h = z & 15;
  const int m0 = blockIdx.y * 128;
  const int n0 = blockIdx.x * 64;
  const _Float16* A = Q  + (size_t)b * TQ_ * D_ + (size_t)h * HD_ + (size_t)m0 * D_;
  const _Float16* W = Kf + (size_t)b * TK_ * D_ + (size_t)h * HD_ + (size_t)n0 * D_;
  v8f acc[4];
#pragma unroll
  for (int j = 0; j < 4; ++j)
#pragma unroll
    for (int i = 0; i < 8; ++i) acc[j][i] = 0.0f;

  mm_tile_128x64<_Float16>(A, D_, W, D_, HD_, Ash, Wsh, acc);

  float* C = S + (size_t)z * TQ_ * TK_;
  const int lane = threadIdx.x & 31;
  const int wave = threadIdx.x >> 5;
  const int mrow = m0 + wave * 16 + ((lane >> 4) << 3);
#pragma unroll
  for (int j = 0; j < 4; ++j) {
    const int n = n0 + j * 16 + (lane & 15);
#pragma unroll
    for (int i = 0; i < 8; ++i)
      C[(size_t)(mrow + i) * TK_ + n] = acc[j][i] * 0.125f;
  }
}

// ---- block reductions -----------------------------------------------------
__device__ inline float block_red_max(float v, float* red) {
  const int tid = threadIdx.x;
  __syncthreads();
  red[tid] = v;
  __syncthreads();
#pragma unroll
  for (int s = 128; s >= 1; s >>= 1) {
    if (tid < s) red[tid] = fmaxf(red[tid], red[tid + s]);
    __syncthreads();
  }
  return red[0];
}
__device__ inline float block_red_sum(float v, float* red) {
  const int tid = threadIdx.x;
  __syncthreads();
  red[tid] = v;
  __syncthreads();
#pragma unroll
  for (int s = 128; s >= 1; s >>= 1) {
    if (tid < s) red[tid] = red[tid] + red[tid + s];
    __syncthreads();
  }
  return red[0];
}

// ---- softmax -> threshold mask (in place) + 2nd-softmax row stats ---------
__global__ __launch_bounds__(256)
void softmax_mask_kernel(float* __restrict__ S, float2* __restrict__ stats,
                         const float* __restrict__ alpha_p) {
  __shared__ float red[256];
  const size_t row = blockIdx.x;          // B*H*TQ rows of length TK
  float* p = S + row * (size_t)TK_;
  const int tid = threadIdx.x;
  const float4 x = ((const float4*)p)[tid];

  float mx = fmaxf(fmaxf(x.x, x.y), fmaxf(x.z, x.w));
  mx = block_red_max(mx, red);
  const float e0 = __expf(x.x - mx), e1 = __expf(x.y - mx);
  const float e2 = __expf(x.z - mx), e3 = __expf(x.w - mx);
  float sm = block_red_sum(e0 + e1 + e2 + e3, red);
  const float inv = 1.0f / sm;
  const float al = alpha_p[0];

  float a0 = e0 * inv; a0 = (a0 >= al) ? a0 : 0.0f;
  float a1 = e1 * inv; a1 = (a1 >= al) ? a1 : 0.0f;
  float a2 = e2 * inv; a2 = (a2 >= al) ? a2 : 0.0f;
  float a3 = e3 * inv; a3 = (a3 >= al) ? a3 : 0.0f;
  ((float4*)p)[tid] = make_float4(a0, a1, a2, a3);

  float mx2 = fmaxf(fmaxf(a0, a1), fmaxf(a2, a3));
  mx2 = block_red_max(mx2, red);
  float s2 = __expf(a0 - mx2) + __expf(a1 - mx2) +
             __expf(a2 - mx2) + __expf(a3 - mx2);
  s2 = block_red_sum(s2, red);
  if (tid == 0) stats[row] = make_float2(mx2, 1.0f / s2);
}

// ---- attn2 @ V: on-the-fly 2nd softmax on A staging, async Vt B copy ------
__global__ __launch_bounds__(256, 2)
void attnv_kernel(const float* __restrict__ S, const float2* __restrict__ stats,
                  const _Float16* __restrict__ Vt, float* __restrict__ Cout) {
  __shared__ alignas(16) _Float16 Ash[2 * ASZ];
  __shared__ alignas(16) _Float16 Wsh[2 * WSZ];
  const int z = blockIdx.z, b = z >> 4, h = z & 15;
  const int m0 = blockIdx.y * 128;
  const int tid = threadIdx.x, lane = tid & 31, wave = tid >> 5;
  const float*    Sb  = S  + (size_t)z * TQ_ * TK_ + (size_t)m0 * TK_;
  const _Float16* Vtb = Vt + (size_t)z * HD_ * TK_;   // [d][k] f16
  const int wr = tid >> 2, wc = (tid & 3) << 3;
  v8f acc[4];
#pragma unroll
  for (int j = 0; j < 4; ++j)
#pragma unroll
    for (int i = 0; i < 8; ++i) acc[j][i] = 0.0f;

  // per-thread chunk coords + row stats (k-invariant -> hoisted)
  int ar[4], ac[4];
  float2 st[4];
#pragma unroll
  for (int i = 0; i < 4; ++i) {
    const int idx = tid + (i << 8);
    ar[i] = idx >> 3;
    ac[i] = (idx & 7) << 2;
    st[i] = stats[(size_t)z * TQ_ + m0 + ar[i]];
  }

  // prologue: stage k0 = 0 into buffer 0
#pragma unroll
  for (int i = 0; i < 4; ++i) {
    const float4 v = *(const float4*)(Sb + (size_t)ar[i] * TK_ + ac[i]);
    h4 o = {(_Float16)(__expf(v.x - st[i].x) * st[i].y),
            (_Float16)(__expf(v.y - st[i].x) * st[i].y),
            (_Float16)(__expf(v.z - st[i].x) * st[i].y),
            (_Float16)(__expf(v.w - st[i].x) * st[i].y)};
    *(h4*)(Ash + ar[i] * LDH + ac[i]) = o;
  }
  async_copy_b128(Vtb + (size_t)wr * TK_ + wc, Wsh + wr * LDH + wc);

  int cur = 0;
#pragma unroll 1
  for (int k0 = 0; k0 < TK_; k0 += 32) {
    _Float16* Ac = Ash + cur * ASZ;
    _Float16* Wc = Wsh + cur * WSZ;
    _Float16* An = Ash + (cur ^ 1) * ASZ;
    _Float16* Wn = Wsh + (cur ^ 1) * WSZ;
    const bool more = (k0 + 32) < TK_;

    async_wait0();
    __syncthreads();

    float4 sreg[4];
    if (more) {
      async_copy_b128(Vtb + (size_t)wr * TK_ + k0 + 32 + wc, Wn + wr * LDH + wc);
#pragma unroll
      for (int i = 0; i < 4; ++i)
        sreg[i] = *(const float4*)(Sb + (size_t)ar[i] * TK_ + k0 + 32 + ac[i]);
    }

    const v16h a = ld_frag16(Ac + (wave * 16 + (lane & 15)) * LDH, lane);
    v16h bf[4];
#pragma unroll
    for (int j = 0; j < 4; ++j)
      bf[j] = ld_frag16(Wc + (j * 16 + (lane & 15)) * LDH, lane);
#pragma unroll
    for (int j = 0; j < 4; ++j)
      acc[j] = __builtin_amdgcn_wmma_f32_16x16x32_f16(false, a, false, bf[j],
                                                      (short)0, acc[j], false, false);

    if (more) {
#pragma unroll
      for (int i = 0; i < 4; ++i) {
        h4 o = {(_Float16)(__expf(sreg[i].x - st[i].x) * st[i].y),
                (_Float16)(__expf(sreg[i].y - st[i].x) * st[i].y),
                (_Float16)(__expf(sreg[i].z - st[i].x) * st[i].y),
                (_Float16)(__expf(sreg[i].w - st[i].x) * st[i].y)};
        *(h4*)(An + ar[i] * LDH + ac[i]) = o;
      }
    }
    cur ^= 1;
  }
  // merge heads: Cout[(b*TQ + q) * D + h*64 + n]
  const int mrow = m0 + wave * 16 + ((lane >> 4) << 3);
#pragma unroll
  for (int j = 0; j < 4; ++j) {
    const int n = j * 16 + (lane & 15);
#pragma unroll
    for (int i = 0; i < 8; ++i)
      Cout[((size_t)b * TQ_ + mrow + i) * D_ + h * HD_ + n] = acc[j][i];
  }
}

// ---- residual add + LayerNorm over D=1024 ---------------------------------
__global__ __launch_bounds__(256)
void add_ln_kernel(const float* __restrict__ a, const float* __restrict__ b,
                   const float* __restrict__ g, const float* __restrict__ beta,
                   float* __restrict__ out) {
  __shared__ float red[256];
  const size_t row = blockIdx.x;
  const int tid = threadIdx.x;
  const float4 va = ((const float4*)(a + row * D_))[tid];
  const float4 vb = ((const float4*)(b + row * D_))[tid];
  const float x0 = va.x + vb.x, x1 = va.y + vb.y;
  const float x2 = va.z + vb.z, x3 = va.w + vb.w;

  const float mu = block_red_sum(x0 + x1 + x2 + x3, red) * (1.0f / D_);
  const float d0 = x0 - mu, d1 = x1 - mu, d2 = x2 - mu, d3 = x3 - mu;
  const float var = block_red_sum(d0 * d0 + d1 * d1 + d2 * d2 + d3 * d3, red)
                    * (1.0f / D_);
  const float rs = rsqrtf(var + 1e-5f);
  const float4 gg = ((const float4*)g)[tid];
  const float4 bt = ((const float4*)beta)[tid];
  ((float4*)(out + row * D_))[tid] =
      make_float4(d0 * rs * gg.x + bt.x, d1 * rs * gg.y + bt.y,
                  d2 * rs * gg.z + bt.z, d3 * rs * gg.w + bt.w);
}

// ---- f32 -> f16 bulk convert (weights) ------------------------------------
__global__ __launch_bounds__(256)
void cvt_f16_kernel(const float* __restrict__ src, _Float16* __restrict__ dst,
                    int n4) {
  const int i = blockIdx.x * 256 + threadIdx.x;
  if (i < n4) {
    const float4 v = ((const float4*)src)[i];
    h4 o = {(_Float16)v.x, (_Float16)v.y, (_Float16)v.z, (_Float16)v.w};
    ((h4*)dst)[i] = o;
  }
}

// ---------------------------------------------------------------------------
extern "C" void kernel_launch(void* const* d_in, const int* in_sizes, int n_in,
                              void* d_out, int out_size, void* d_ws, size_t ws_size,
                              hipStream_t stream) {
  const float* base   = (const float*)d_in[0];
  const float* fusion = (const float*)d_in[1];
  const float* g1 = (const float*)d_in[6],  * b1 = (const float*)d_in[7];
  const float* g2 = (const float*)d_in[8],  * b2 = (const float*)d_in[9];
  const float* fc1_b = (const float*)d_in[11];
  const float* fc2_b = (const float*)d_in[13];
  const float* rb    = (const float*)d_in[15];
  const float* alpha = (const float*)d_in[16];

  const size_t NE = (size_t)B_ * TQ_ * D_;   // 8M elems per activation
  const size_t WN = (size_t)D_ * D_;         // 1M elems per weight
  float* out_final = (float*)d_out;          // [B,TQ,D]
  float* Smask     = out_final + NE;         // [B,H,TQ,TK] (scratch -> masked)

  // workspace layout: f16 weights(7) | Qh | Kh | Vt | 4x f32 bufs | stats
  _Float16* wsh = (_Float16*)d_ws;
  _Float16* W16[7];
  for (int i = 0; i < 7; ++i) W16[i] = wsh + (size_t)i * WN;
  _Float16* Qh = wsh + 7 * WN;
  _Float16* Kh = Qh + NE;
  _Float16* Vt = Kh + NE;                    // [B,H,HD,TK] f16 (V transposed)
  float* f32b = (float*)(Vt + NE);
  float* out_attn = f32b;                    // -> later reused as y
  float* tmp      = f32b + NE;               // Wo out -> later t2
  float* x        = f32b + 2 * NE;
  float* T1       = f32b + 3 * NE;
  float2* stats   = (float2*)(f32b + 4 * NE);
  float* y        = out_attn;

  const dim3 blk(256);
  const dim3 gD(D_ / 64, (B_ * TQ_) / 128);  // 16 x 64 CTAs for D-GEMMs
  const dim3 gCvt((int)(WN / 4 / 256));      // 1024 blocks per weight

  // weight dict slots: Wq=2 Wk=3 Wv=4 Wo=5 fc1_w=10 fc2_w=12 rw=14
  const int wslot[7] = {2, 3, 4, 5, 10, 12, 14};
  for (int i = 0; i < 7; ++i)
    cvt_f16_kernel<<<gCvt, blk, 0, stream>>>((const float*)d_in[wslot[i]],
                                             W16[i], (int)(WN / 4));

  // Q/K/V projections (Q,K -> f16 row-major; V -> f16 transposed per head)
  gemm_epi_kernel<<<gD, blk, 0, stream>>>(base,   D_, W16[0], D_, nullptr, Qh, D_, D_, 0, 1);
  gemm_epi_kernel<<<gD, blk, 0, stream>>>(fusion, D_, W16[1], D_, nullptr, Kh, D_, D_, 0, 1);
  gemm_epi_kernel<<<gD, blk, 0, stream>>>(fusion, D_, W16[2], D_, nullptr, Vt, D_, D_, 0, 2);

  // scores -> (in-place) softmax+mask -> attn2 @ V
  scores_kernel<<<dim3(TK_ / 64, TQ_ / 128, B_ * H_), blk, 0, stream>>>(Qh, Kh, Smask);
  softmax_mask_kernel<<<dim3(B_ * H_ * TQ_), blk, 0, stream>>>(Smask, stats, alpha);
  attnv_kernel<<<dim3(1, TQ_ / 128, B_ * H_), blk, 0, stream>>>(Smask, stats, Vt, out_attn);

  // output proj + LN1
  gemm_epi_kernel<<<gD, blk, 0, stream>>>(out_attn, D_, W16[3], D_, nullptr, tmp, D_, D_, 0, 0);
  add_ln_kernel<<<dim3(B_ * TQ_), blk, 0, stream>>>(base, tmp, g1, b1, x);

  // FFN: silu(silu(x@fc1^T+b)@fc2^T+b) + LN2
  gemm_epi_kernel<<<gD, blk, 0, stream>>>(x,  D_, W16[4], D_, fc1_b, T1,  D_, D_, 1, 0);
  gemm_epi_kernel<<<gD, blk, 0, stream>>>(T1, D_, W16[5], D_, fc2_b, tmp, D_, D_, 1, 0);
  add_ln_kernel<<<dim3(B_ * TQ_), blk, 0, stream>>>(x, tmp, g2, b2, y);

  // final projection
  gemm_epi_kernel<<<gD, blk, 0, stream>>>(y, D_, W16[6], D_, rb, out_final, D_, D_, 0, 0);
}
